// MambaEncoder_4896262717622
// MI455X (gfx1250) — matrix operand order repaired and checked by reading, not compiled, not run on hardware
//
#include <hip/hip_runtime.h>
#include <hip/hip_bf16.h>

// ---------------------------------------------------------------------------
// Mamba encoder for MI455X (gfx1250, wave32, WMMA, async-to-LDS).
//   B=8, T=2048, IN=32, HID=256, OUT=128, NL=3, D_STATE=16, D_CONV=4,
//   D_INNER=512, DT_RANK=16, M = B*T = 16384 token-rows.
// f16 activation/weight storage (f32 accumulation); big GEMMs use
// double-buffered async-to-LDS weight staging, 2x4 WMMA tiles per wave.
// ---------------------------------------------------------------------------

#define B_      8
#define T_      2048
#define IN_DIM  32
#define HID     256
#define OUT_DIM 128
#define NL      3
#define D_STATE 16
#define D_CONV  4
#define D_INNER 512
#define DT_RANK 16
#define MROWS   (B_ * T_)

typedef __attribute__((ext_vector_type(16))) _Float16 v16h;
typedef __attribute__((ext_vector_type(8)))  _Float16 v8h;
typedef __attribute__((ext_vector_type(8)))  float    v8f;

typedef __attribute__((address_space(3))) _Float16 lds_f16_t;

__device__ __forceinline__ float silu_f(float x) {
    return x / (1.f + __expf(-x));
}
__device__ __forceinline__ float softplus_f(float x) {
    return (x > 20.f) ? x : log1pf(__expf(x));
}

__device__ __forceinline__ void async_fill_b128(unsigned ldsoff,
                                                const _Float16* gsrc)
{
    const unsigned long long gp = (unsigned long long)(size_t)gsrc;
    asm volatile("global_load_async_to_lds_b128 %0, %1, off"
                 :: "v"(ldsoff), "v"(gp) : "memory");
}
__device__ __forceinline__ void wait_async0()
{
    asm volatile("s_wait_asynccnt 0x0" ::: "memory");
}

// ---------------------------------------------------------------------------
// f32 -> f16 conversion (weights + network input), run once per launch.
// ---------------------------------------------------------------------------
__global__ void cvt_f32_to_f16(const float* __restrict__ src,
                               _Float16* __restrict__ dst, long n)
{
    const long i = (long)blockIdx.x * blockDim.x + threadIdx.x;
    if (i < n) dst[i] = (_Float16)src[i];
}

// ---------------------------------------------------------------------------
// LDS-staged WMMA GEMM:  O[M,N] = X[M,K](lda) * W[N,K](ldw)^T (+bias)
// Requirements: K % 32 == 0, N % 64 == 0, M % 256 == 0.
// Block = 256 threads (8 waves). Block tile = 256 rows x 64 cols
// (each wave: 2 row-tiles x 4 col-tiles = 8 WMMAs per K-step).
// Per K-step the 4 shared 16x32 B-tiles (4 KB) are DMA'd into a
// double-buffered LDS region with global_load_async_to_lds_b128
// (one b128 per thread), hidden behind the previous step's compute.
// ---------------------------------------------------------------------------
__global__ void gemm_wmma_lds(const _Float16* __restrict__ X, int lda,
                              const _Float16* __restrict__ W, int ldw,
                              _Float16* __restrict__ O, int ldo,
                              const float* __restrict__ bias, int K)
{
    __shared__ _Float16 ldsB[4096];           // 2 buffers * 4 tiles * 512 halves

    const int tid  = threadIdx.x;
    const int lane = tid & 31;
    const int wave = tid >> 5;
    const int row0 = (blockIdx.x * 8 + wave) * 32;   // 2 row-tiles per wave
    const int col0 = blockIdx.y * 64;
    const int lrow = lane & 15;
    const bool hi  = lane >= 16;

    // ISA A-layout (16-bit 16x32): lanes 0-15 hold K 0-7 / 16-23,
    //                              lanes 16-31 hold K 8-15 / 24-31.
    const int a_off0 = hi ? 8  : 0;
    const int a_off1 = hi ? 24 : 16;

    // Async fill mapping: thread tid moves one 16-byte chunk per K-step.
    // LDS layout == the B-fragment register layout read back below.
    const int fc = tid >> 6;                  // tile 0..3
    const int fl = (tid & 63) >> 1;           // frag lane 0..31
    const int fp = tid & 1;                   // k-half 0/1
    const int fn = col0 + fc * 16 + (fl & 15);
    const int fk = ((fl < 16) ? 0 : 8) + fp * 16;
    const _Float16* __restrict__ wsrc = W + (size_t)fn * ldw + fk;
    const unsigned ldsoff0 =
        (unsigned)(size_t)((lds_f16_t*)ldsB + (size_t)tid * 8);
    const unsigned ldsoff1 = ldsoff0 + 4096;  // bytes: second buffer

    const _Float16* __restrict__ xrow0 = X + (size_t)(row0 + lrow) * lda;
    const _Float16* __restrict__ xrow1 = X + (size_t)(row0 + 16 + lrow) * lda;

    v8f acc[2][4] = {{{}, {}, {}, {}}, {{}, {}, {}, {}}};

    // Prologue: DMA first B-tile set into buffer 0.
    async_fill_b128(ldsoff0, wsrc);

    int ibuf = 0;
    for (int kk = 0; kk < K; kk += 32, ibuf ^= 1) {
        wait_async0();           // my chunk of the current buffer has landed
        __syncthreads();         // everyone's chunks landed AND the other
                                 // buffer's readers (last step) are done
        if (kk + 32 < K)         // kick off next K-tile DMA (other buffer)
            async_fill_b128(ibuf ? ldsoff0 : ldsoff1, wsrc + kk + 32);

        // A fragments (2 row-tiles) from global.
        v16h a0, a1;
        {
            const v8h l0 = *(const v8h*)(xrow0 + kk + a_off0);
            const v8h h0 = *(const v8h*)(xrow0 + kk + a_off1);
            const v8h l1 = *(const v8h*)(xrow1 + kk + a_off0);
            const v8h h1 = *(const v8h*)(xrow1 + kk + a_off1);
#pragma unroll
            for (int i = 0; i < 8; ++i) {
                a0[i] = l0[i]; a0[8 + i] = h0[i];
                a1[i] = l1[i]; a1[8 + i] = h1[i];
            }
        }

        // All 4 B fragments up front (8 x ds_load_b128, one dscnt wait).
        const _Float16* bbase = &ldsB[ibuf ? 2048 : 0];
        v16h bf[4];
#pragma unroll
        for (int c = 0; c < 4; ++c) {
            const _Float16* bp = bbase + (c * 32 + lane) * 16;
            const v8h b0 = *(const v8h*)bp;
            const v8h b1 = *(const v8h*)(bp + 8);
#pragma unroll
            for (int i = 0; i < 8; ++i) { bf[c][i] = b0[i]; bf[c][8 + i] = b1[i]; }
        }

        // 8 back-to-back WMMAs (B reused across both row-tiles).
#pragma unroll
        for (int c = 0; c < 4; ++c) {
            acc[0][c] = __builtin_amdgcn_wmma_f32_16x16x32_f16(
                            false, a0, false, bf[c], (short)0, acc[0][c],
                            false, false);
            acc[1][c] = __builtin_amdgcn_wmma_f32_16x16x32_f16(
                            false, a1, false, bf[c], (short)0, acc[1][c],
                            false, false);
        }
    }

    // C/D layout: VGPR r -> row r (+8 for hi lanes), lane&15 -> column.
#pragma unroll
    for (int c = 0; c < 4; ++c) {
        const int   col = col0 + c * 16 + lrow;
        const float bv  = bias ? bias[col] : 0.f;
#pragma unroll
        for (int rt = 0; rt < 2; ++rt) {
#pragma unroll
            for (int r = 0; r < 8; ++r) {
                const int row = row0 + rt * 16 + r + (hi ? 8 : 0);
                O[(size_t)row * ldo + col] = (_Float16)(acc[rt][c][r] + bv);
            }
        }
    }
}

// ---------------------------------------------------------------------------
// Simple WMMA GEMM (one 16x16 tile per wave) for the skinny cases:
//   x_proj (N=48), dt_proj (K=16, bias + softplus).
// Branch-free main loop + zero-padded tail. act: 0=identity, 1=softplus.
// ---------------------------------------------------------------------------
__global__ void gemm_wmma_simple(const _Float16* __restrict__ X, int lda,
                                 const _Float16* __restrict__ W, int ldw,
                                 _Float16* __restrict__ O, int ldo,
                                 const float* __restrict__ bias,
                                 int K, int act)
{
    const int lane = threadIdx.x & 31;
    const int wave = threadIdx.x >> 5;
    const int row0 = (blockIdx.x * 8 + wave) * 16;
    const int col0 = blockIdx.y * 16;
    const int lrow = lane & 15;
    const bool hi  = lane >= 16;

    const int a_off0 = hi ? 16 : 0;
    const int a_off1 = hi ? 24 : 8;
    const int b_off  = hi ? 16 : 0;

    const _Float16* __restrict__ xrow = X + (size_t)(row0 + lrow) * lda;
    const _Float16* __restrict__ wrow = W + (size_t)(col0 + lrow) * ldw;

    v8f acc = {};
    int kk = 0;
    for (; kk + 32 <= K; kk += 32) {          // branch-free main loop
        v16h a, bm;
        const v8h a0 = *(const v8h*)(xrow + kk + a_off0);
        const v8h a1 = *(const v8h*)(xrow + kk + a_off1);
        const v8h b0 = *(const v8h*)(wrow + kk + b_off);
        const v8h b1 = *(const v8h*)(wrow + kk + b_off + 8);
#pragma unroll
        for (int i = 0; i < 8; ++i) {
            a[i] = a0[i];  a[8 + i] = a1[i];
            bm[i] = b0[i]; bm[8 + i] = b1[i];
        }
        if (kk + 64 <= K) {
            __builtin_prefetch(xrow + kk + 32 + a_off0, 0, 1);
            __builtin_prefetch(wrow + kk + 32 + b_off,  0, 1);
        }
        acc = __builtin_amdgcn_wmma_f32_16x16x32_f16(
                  false, a, false, bm, (short)0, acc, false, false);
    }
    if (kk < K) {                             // zero-padded tail (K=16 case)
        v16h a, bm;
#pragma unroll
        for (int i = 0; i < 8; ++i) {
            const int k0 = kk + a_off0 + i;
            const int k1 = kk + a_off1 + i;
            a[i]     = (k0 < K) ? xrow[k0] : (_Float16)0.f;
            a[8 + i] = (k1 < K) ? xrow[k1] : (_Float16)0.f;
        }
#pragma unroll
        for (int i = 0; i < 16; ++i) {
            const int k = kk + b_off + i;
            bm[i] = (k < K) ? wrow[k] : (_Float16)0.f;
        }
        acc = __builtin_amdgcn_wmma_f32_16x16x32_f16(
                  false, a, false, bm, (short)0, acc, false, false);
    }

    const int   col = col0 + lrow;
    const float bv  = bias ? bias[col] : 0.f;
#pragma unroll
    for (int r = 0; r < 8; ++r) {
        const int row = row0 + r + (hi ? 8 : 0);
        float v = acc[r] + bv;
        if (act == 1) v = softplus_f(v);
        O[(size_t)row * ldo + col] = (_Float16)v;
    }
}

// ---------------------------------------------------------------------------
// Causal depthwise conv (K=4) + bias + SiLU on the u-half of xz (f16 I/O).
// ---------------------------------------------------------------------------
__global__ void conv_silu_kernel(const _Float16* __restrict__ xz,  // [M,1024]
                                 const float* __restrict__ cw,     // [512,4]
                                 const float* __restrict__ cb,     // [512]
                                 _Float16* __restrict__ u)         // [M,512]
{
    const long idx = (long)blockIdx.x * blockDim.x + threadIdx.x;  // M*512
    const int  d   = (int)(idx & (D_INNER - 1));
    const long row = idx >> 9;                                     // b*T + t
    const int  t   = (int)(row & (T_ - 1));
    float acc = cb[d];
#pragma unroll
    for (int k = 0; k < D_CONV; ++k) {
        const int tt = t - (D_CONV - 1) + k;
        if (tt >= 0)
            acc += (float)xz[(row - (D_CONV - 1) + k) * (2 * D_INNER) + d]
                   * cw[d * D_CONV + k];
    }
    u[idx] = (_Float16)silu_f(acc);
}

// ---------------------------------------------------------------------------
// Selective scan: one thread per (b, d); 16 f32 states in VGPRs.
// B/C vectors (xdbl cols 16..47) are lane-uniform loads.
// Grid: (D_INNER/256, B), block 256.
// ---------------------------------------------------------------------------
__global__ void scan_kernel(const _Float16* __restrict__ delta, // [M,512]
                            const _Float16* __restrict__ u,     // [M,512]
                            const _Float16* __restrict__ xdbl,  // [M,48]
                            const _Float16* __restrict__ xz,    // [M,1024]
                            const float* __restrict__ A_log,    // [512,16]
                            const float* __restrict__ Dp,       // [512]
                            _Float16* __restrict__ ys)          // [M,512]
{
    const int d = blockIdx.x * blockDim.x + threadIdx.x;
    const int b = blockIdx.y;

    float A[D_STATE];
#pragma unroll
    for (int s = 0; s < D_STATE; ++s) A[s] = -__expf(A_log[d * D_STATE + s]);
    const float Dd = Dp[d];

    float h[D_STATE];
#pragma unroll
    for (int s = 0; s < D_STATE; ++s) h[s] = 0.f;

    for (int t = 0; t < T_; ++t) {
        const size_t row = (size_t)b * T_ + t;
        const float  de  = (float)delta[row * D_INNER + d];
        const float  uu  = (float)u[row * D_INNER + d];
        const float  du  = de * uu;
        const _Float16* bc = xdbl + row * (DT_RANK + 2 * D_STATE);
        float y = 0.f;
#pragma unroll
        for (int s = 0; s < D_STATE; ++s) {
            const float dA = __expf(de * A[s]);
            h[s] = dA * h[s] + du * (float)bc[DT_RANK + s];          // B_t
            y   += h[s] * (float)bc[DT_RANK + D_STATE + s];          // C_t
        }
        y += uu * Dd;
        const float z = (float)xz[row * (2 * D_INNER) + D_INNER + d];
        ys[row * D_INNER + d] = (_Float16)(y * silu_f(z));
    }
}

// ---------------------------------------------------------------------------
// Mean pool over T (f16 in, f32 out). Grid: (B), block: HID.
// ---------------------------------------------------------------------------
__global__ void pool_kernel(const _Float16* __restrict__ hbuf,
                            float* __restrict__ pooled)
{
    const int hcol = threadIdx.x;
    const int b    = blockIdx.x;
    float acc = 0.f;
    for (int t = 0; t < T_; ++t)
        acc += (float)hbuf[((size_t)b * T_ + t) * HID + hcol];
    pooled[b * HID + hcol] = acc * (1.f / (float)T_);
}

// ---------------------------------------------------------------------------
// Final projection (f32): out[b,o] = pooled[b,:] . W[o,:] + bias[o]
// ---------------------------------------------------------------------------
__global__ void outproj_kernel(const float* __restrict__ pooled,
                               const float* __restrict__ W,
                               const float* __restrict__ bias,
                               float* __restrict__ out)
{
    const int idx = blockIdx.x * blockDim.x + threadIdx.x;
    if (idx >= B_ * OUT_DIM) return;
    const int b = idx / OUT_DIM;
    const int o = idx % OUT_DIM;
    float acc = bias[o];
    for (int h = 0; h < HID; ++h)
        acc += pooled[b * HID + h] * W[o * HID + h];
    out[idx] = acc;
}

// ---------------------------------------------------------------------------
extern "C" void kernel_launch(void* const* d_in, const int* in_sizes, int n_in,
                              void* d_out, int out_size, void* d_ws, size_t ws_size,
                              hipStream_t stream)
{
    (void)in_sizes; (void)n_in; (void)out_size; (void)ws_size;

    const float* x    = (const float*)d_in[0];   // [8,2048,32]
    const float* ipw  = (const float*)d_in[1];   // [256,32]
    const float* ipb  = (const float*)d_in[2];   // [256]
    const float* inw  = (const float*)d_in[3];   // [3,1024,256]
    const float* cw   = (const float*)d_in[4];   // [3,512,4]
    const float* cb   = (const float*)d_in[5];   // [3,512]
    const float* xw   = (const float*)d_in[6];   // [3,48,512]
    const float* dtw  = (const float*)d_in[7];   // [3,512,16]
    const float* dtb  = (const float*)d_in[8];   // [3,512]
    const float* Alog = (const float*)d_in[9];   // [3,512,16]
    const float* Dpar = (const float*)d_in[10];  // [3,512]
    const float* ow   = (const float*)d_in[11];  // [3,256,512]
    const float* opw  = (const float*)d_in[12];  // [128,256]
    const float* opb  = (const float*)d_in[13];  // [128]
    float*       out  = (float*)d_out;           // [8,128]

    // Workspace carve-up (halves; every size is a multiple of 8 -> 16B align).
    _Float16* p = (_Float16*)d_ws;
    _Float16* xh    = p; p += (size_t)MROWS * IN_DIM;            // f16 input
    _Float16* wip   = p; p += (size_t)HID * IN_DIM;
    _Float16* winp  = p; p += (size_t)NL * 2 * D_INNER * HID;
    _Float16* wxp   = p; p += (size_t)NL * 48 * D_INNER;
    _Float16* wdt   = p; p += (size_t)NL * D_INNER * DT_RANK;
    _Float16* wowp  = p; p += (size_t)NL * HID * D_INNER;
    _Float16* hbuf  = p; p += (size_t)MROWS * HID;               // [M,256]
    _Float16* xzbuf = p; p += (size_t)MROWS * 2 * D_INNER;       // [M,1024]
    _Float16* ubuf  = p; p += (size_t)MROWS * D_INNER;           // [M,512]
    _Float16* xdbl  = p; p += (size_t)MROWS * 48;                // [M,48]
    _Float16* delt  = p; p += (size_t)MROWS * D_INNER;           // [M,512]
    _Float16* ysb   = p; p += (size_t)MROWS * D_INNER;           // [M,512]
    float*    pooled = (float*)p;                                // [8,256]

    const dim3 blk(256);
    auto cvt = [&](const float* s, _Float16* d, long n) {
        cvt_f32_to_f16<<<(unsigned)((n + 255) / 256), blk, 0, stream>>>(s, d, n);
    };
    cvt(x,   xh,   (long)MROWS * IN_DIM);
    cvt(ipw, wip,  (long)HID * IN_DIM);
    cvt(inw, winp, (long)NL * 2 * D_INNER * HID);
    cvt(xw,  wxp,  (long)NL * 48 * D_INNER);
    cvt(dtw, wdt,  (long)NL * D_INNER * DT_RANK);
    cvt(ow,  wowp, (long)NL * HID * D_INNER);

    // h = x @ input_proj_w^T + b            (M x 256, K=32)
    gemm_wmma_lds<<<dim3(MROWS / 256, HID / 64), blk, 0, stream>>>(
        xh, IN_DIM, wip, IN_DIM, hbuf, HID, ipb, IN_DIM);

    for (int l = 0; l < NL; ++l) {
        const _Float16* winp_l = winp + (size_t)l * 2 * D_INNER * HID;
        const _Float16* wxp_l  = wxp  + (size_t)l * 48 * D_INNER;
        const _Float16* wdt_l  = wdt  + (size_t)l * D_INNER * DT_RANK;
        const _Float16* wow_l  = wowp + (size_t)l * HID * D_INNER;
        const float* cw_l   = cw   + (size_t)l * D_INNER * D_CONV;
        const float* cb_l   = cb   + (size_t)l * D_INNER;
        const float* dtb_l  = dtb  + (size_t)l * D_INNER;
        const float* Alog_l = Alog + (size_t)l * D_INNER * D_STATE;
        const float* Dpar_l = Dpar + (size_t)l * D_INNER;

        // xz = h @ in_proj_w^T               (M x 1024, K=256)
        gemm_wmma_lds<<<dim3(MROWS / 256, (2 * D_INNER) / 64), blk, 0, stream>>>(
            hbuf, HID, winp_l, HID, xzbuf, 2 * D_INNER, nullptr, HID);

        // u = silu(causal_dwconv(xz[:, :512]))
        conv_silu_kernel<<<(MROWS * D_INNER) / 256, blk, 0, stream>>>(
            xzbuf, cw_l, cb_l, ubuf);

        // x_dbl = u @ x_proj_w^T             (M x 48, K=512)
        gemm_wmma_simple<<<dim3(MROWS / 128, 48 / 16), blk, 0, stream>>>(
            ubuf, D_INNER, wxp_l, D_INNER, xdbl, 48, nullptr, D_INNER, 0);

        // delta = softplus(dt @ dt_proj_w^T + dtb)   (M x 512, K=16)
        gemm_wmma_simple<<<dim3(MROWS / 128, D_INNER / 16), blk, 0, stream>>>(
            xdbl, 48, wdt_l, DT_RANK, delt, D_INNER, dtb_l, DT_RANK, 1);

        // selective scan + gate
        scan_kernel<<<dim3(D_INNER / 256, B_), blk, 0, stream>>>(
            delt, ubuf, xdbl, xzbuf, Alog_l, Dpar_l, ysb);

        // h = ys @ out_proj_w^T              (M x 256, K=512)
        gemm_wmma_lds<<<dim3(MROWS / 256, HID / 64), blk, 0, stream>>>(
            ysb, D_INNER, wow_l, D_INNER, hbuf, HID, nullptr, D_INNER);
    }

    pool_kernel<<<B_, HID, 0, stream>>>(hbuf, pooled);
    outproj_kernel<<<(B_ * OUT_DIM + 255) / 256, blk, 0, stream>>>(
        pooled, opw, opb, out);
}
